// PointTransformerBlock_11725260718338
// MI455X (gfx1250) — compile-verified, matrix operand order checked by reading, not run on hardware
//
#include <hip/hip_runtime.h>
#include <hip/hip_bf16.h>
#include <math.h>

// ---------------------------------------------------------------------------
// Point Transformer block, fused for MI455X (gfx1250, wave32, WMMA).
// Shapes fixed by setup_inputs: B=2, N=8192, K=16, d=64.
// All 64x64 channel matmuls use v_wmma_f32_16x16x32_f16 (f16 in, f32 acc).
// BN scale g/sqrt(1+eps) folded into f16 weights once per launch.
// ---------------------------------------------------------------------------

typedef _Float16 v16h __attribute__((ext_vector_type(16)));
typedef _Float16 v8h  __attribute__((ext_vector_type(8)));
typedef float    v8f  __attribute__((ext_vector_type(8)));

#define BB 2
#define NN 8192
#define KK 16
#define DD 64
#define NEGS 0.2f
#define TSTRIDE 72   // LDS tile column stride in halves (bank-conflict padding)

__device__ __forceinline__ float lrelu(float x) { return x > 0.f ? x : NEGS * x; }

__device__ __forceinline__ v16h cat16(v8h lo, v8h hi) {
  v16h r;
#pragma unroll
  for (int i = 0; i < 8; i++) { r[i] = lo[i]; r[i + 8] = hi[i]; }
  return r;
}

// D = A x B + C, 16x16x32 f16 -> f32
__device__ __forceinline__ v8f wmma_f16(v16h a, v16h b, v8f c) {
  return __builtin_amdgcn_wmma_f32_16x16x32_f16(false, a, false, b, (short)0, c,
                                                false, false);
}

// A-fragment from row-major 64x64 f16 weight. Rows = 16*rb + nl, k-step s.
// Per-lane K map: j<8 -> K=32s+8h+j ; j>=8 -> K=32s+16+8h+(j-8).
__device__ __forceinline__ v16h ldA(const _Float16* W, int rb, int s, int nl, int h) {
  const int row = 16 * rb + nl;
  v8h lo = *(const v8h*)(W + row * 64 + 32 * s + 8 * h);
  v8h hi = *(const v8h*)(W + row * 64 + 32 * s + 16 + 8 * h);
  return cat16(lo, hi);
}

// B-fragment from LDS tile (column-major, TSTRIDE halves per column).
__device__ __forceinline__ v16h ldB_lds(const _Float16* t, int s, int nl, int h) {
  v8h lo = *(const v8h*)(t + nl * TSTRIDE + 32 * s + 8 * h);
  v8h hi = *(const v8h*)(t + nl * TSTRIDE + 32 * s + 16 + 8 * h);
  return cat16(lo, hi);
}

// ---------------------------------------------------------------------------
// Kernel 0: fold BN scales into weights, convert to f16 workspace copies.
// ---------------------------------------------------------------------------
__global__ void prep_kernel(const float* W1, const float* g1,
                            const float* Wq, const float* Wk, const float* Wv,
                            const float* Wd1, const float* gd1,
                            const float* Wd2, const float* gd2,
                            const float* Wg1, const float* gg1,
                            const float* Wg2, const float* gg2,
                            const float* W2, const float* g2,
                            _Float16* hW1, _Float16* hWq, _Float16* hWk,
                            _Float16* hWv, _Float16* hWd2, _Float16* hWg1,
                            _Float16* hWg2, float* fWd1s, float* fW2s) {
  const float inv = rsqrtf(1.0f + 1e-5f);
  int tid = blockIdx.x * blockDim.x + threadIdx.x;
  int nthr = gridDim.x * blockDim.x;
  for (int i = tid; i < 4096; i += nthr) {
    int m = i >> 6;
    hW1[i]  = (_Float16)(W1[i] * g1[m] * inv);
    hWq[i]  = (_Float16)Wq[i];
    hWk[i]  = (_Float16)Wk[i];
    hWv[i]  = (_Float16)Wv[i];
    hWd2[i] = (_Float16)(Wd2[i] * gd2[m] * inv);
    hWg1[i] = (_Float16)(Wg1[i] * gg1[m] * inv);
    hWg2[i] = (_Float16)(Wg2[i] * gg2[m] * inv);
    fW2s[i] = W2[i] * g2[m] * inv;
  }
  for (int i = tid; i < 192; i += nthr) {
    int m = i / 3;
    fWd1s[i] = Wd1[i] * gd1[m] * inv;
  }
}

// ---------------------------------------------------------------------------
// Kernel 1: x = lrelu(bn(W1*feats)); q/k/v = Wq/Wk/Wv * x.
// One wave per 16-point column tile; 4 waves per block.
// q -> [B,64,N] f32 ; k,v -> [B,N,64] f16 (gather-friendly).
// ---------------------------------------------------------------------------
__global__ __launch_bounds__(128) void qkv_kernel(
    const float* __restrict__ feats, const float* __restrict__ b1,
    const _Float16* __restrict__ hW1, const _Float16* __restrict__ hWq,
    const _Float16* __restrict__ hWk, const _Float16* __restrict__ hWv,
    float* __restrict__ q, _Float16* __restrict__ kf, _Float16* __restrict__ vf) {
  __shared__ _Float16 xt[4][16 * TSTRIDE];
  const int lane = threadIdx.x & 31, w = threadIdx.x >> 5;
  const int nl = lane & 15, h = lane >> 4;
  const int b = blockIdx.z;
  const int n0 = (blockIdx.x * 4 + w) * 16;

  // B-fragments of feats (f32 -> f16 on the fly)
  v16h bf[2];
#pragma unroll
  for (int s = 0; s < 2; s++) {
    v16h t;
#pragma unroll
    for (int j = 0; j < 16; j++) {
      int K = 32 * s + ((j >> 3) << 4) + (h << 3) + (j & 7);
      t[j] = (_Float16)feats[(b * DD + K) * NN + n0 + nl];
    }
    bf[s] = t;
  }

  // x = lrelu(bn(W1 * feats)) -> LDS tile (column-major f16)
#pragma unroll
  for (int rb = 0; rb < 4; rb++) {
    v8f acc = {};
    acc = wmma_f16(ldA(hW1, rb, 0, nl, h), bf[0], acc);
    acc = wmma_f16(ldA(hW1, rb, 1, nl, h), bf[1], acc);
#pragma unroll
    for (int r = 0; r < 8; r++) {
      int m = 16 * rb + r + 8 * h;
      xt[w][nl * TSTRIDE + m] = (_Float16)lrelu(acc[r] + b1[m]);
    }
  }
  __syncthreads();

  v16h bx0 = ldB_lds(xt[w], 0, nl, h);
  v16h bx1 = ldB_lds(xt[w], 1, nl, h);

  // q (store [B,64,N] f32, lanes 0..15 coalesced over n)
#pragma unroll
  for (int rb = 0; rb < 4; rb++) {
    v8f acc = {};
    acc = wmma_f16(ldA(hWq, rb, 0, nl, h), bx0, acc);
    acc = wmma_f16(ldA(hWq, rb, 1, nl, h), bx1, acc);
#pragma unroll
    for (int r = 0; r < 8; r++) {
      int m = 16 * rb + r + 8 * h;
      q[(b * DD + m) * NN + n0 + nl] = acc[r];
    }
  }
  // k (store point-major f16, one b128 store per rb)
#pragma unroll
  for (int rb = 0; rb < 4; rb++) {
    v8f acc = {};
    acc = wmma_f16(ldA(hWk, rb, 0, nl, h), bx0, acc);
    acc = wmma_f16(ldA(hWk, rb, 1, nl, h), bx1, acc);
    v8h o;
#pragma unroll
    for (int r = 0; r < 8; r++) o[r] = (_Float16)acc[r];
    *(v8h*)(kf + ((size_t)(b * NN + n0 + nl)) * DD + 16 * rb + 8 * h) = o;
  }
  // v
#pragma unroll
  for (int rb = 0; rb < 4; rb++) {
    v8f acc = {};
    acc = wmma_f16(ldA(hWv, rb, 0, nl, h), bx0, acc);
    acc = wmma_f16(ldA(hWv, rb, 1, nl, h), bx1, acc);
    v8h o;
#pragma unroll
    for (int r = 0; r < 8; r++) o[r] = (_Float16)acc[r];
    *(v8h*)(vf + ((size_t)(b * NN + n0 + nl)) * DD + 16 * rb + 8 * h) = o;
  }
}

// ---------------------------------------------------------------------------
// Kernel 2: kNN (k=16, includes self). Thread-per-query, 256-candidate LDS
// chunks, register-resident top-16 via unrolled insertion (order-free: the
// attention is permutation-invariant over neighbors).
// ---------------------------------------------------------------------------
__global__ __launch_bounds__(256) void knn_kernel(const float* __restrict__ pos,
                                                  int* __restrict__ idxo) {
  __shared__ float cx[256], cy[256], cz[256];
  const int t = threadIdx.x;
  const int b = blockIdx.z;
  const int n = blockIdx.x * 256 + t;
  const float qx = pos[(b * 3 + 0) * NN + n];
  const float qy = pos[(b * 3 + 1) * NN + n];
  const float qz = pos[(b * 3 + 2) * NN + n];

  float bd[KK];
  int bi[KK];
#pragma unroll
  for (int s = 0; s < KK; s++) { bd[s] = 3.0e38f; bi[s] = 0; }
  float worst = 3.0e38f;
  int wslot = 0;

  for (int c0 = 0; c0 < NN; c0 += 256) {
    __syncthreads();
    cx[t] = pos[(b * 3 + 0) * NN + c0 + t];
    cy[t] = pos[(b * 3 + 1) * NN + c0 + t];
    cz[t] = pos[(b * 3 + 2) * NN + c0 + t];
    __syncthreads();
    if (c0 + 256 < NN) {  // warm L2/GL1 for the next chunk (global_prefetch_b8)
      __builtin_prefetch(&pos[(b * 3 + 0) * NN + c0 + 256 + t], 0, 0);
      __builtin_prefetch(&pos[(b * 3 + 1) * NN + c0 + 256 + t], 0, 0);
      __builtin_prefetch(&pos[(b * 3 + 2) * NN + c0 + 256 + t], 0, 0);
    }
#pragma unroll 4
    for (int c = 0; c < 256; c++) {
      float dx = qx - cx[c], dy = qy - cy[c], dz = qz - cz[c];
      float d = dx * dx + dy * dy + dz * dz;
      if (d < worst) {
#pragma unroll
        for (int s = 0; s < KK; s++)
          if (s == wslot) { bd[s] = d; bi[s] = c0 + c; }
        worst = -1.0f;
#pragma unroll
        for (int s = 0; s < KK; s++)
          if (bd[s] > worst) { worst = bd[s]; wslot = s; }
      }
    }
  }
#pragma unroll
  for (int s = 0; s < KK; s++)
    idxo[((size_t)(b * NN + n)) * KK + s] = bi[s];
}

// ---------------------------------------------------------------------------
// Kernel 3: fully fused attention tail, wave-per-point.
// pos_enc (Wd1 VALU + Wd2 WMMA) -> a = q-k+pe -> Wg1 -> Wg2 (WMMA, LDS
// tile round-trips) -> softmax over K -> sum attn*(v+pe) -> W2 matvec +
// residual. No [B,64,K,N] tensor ever touches HBM.
// ---------------------------------------------------------------------------
__global__ __launch_bounds__(128) void attn_kernel(
    const float* __restrict__ feats, const float* __restrict__ pos,
    const float* __restrict__ qg, const _Float16* __restrict__ kf,
    const _Float16* __restrict__ vf, const int* __restrict__ knn,
    const _Float16* __restrict__ hWd2, const _Float16* __restrict__ hWg1,
    const _Float16* __restrict__ hWg2, const float* __restrict__ fWd1s,
    const float* __restrict__ fW2s, const float* __restrict__ bd1,
    const float* __restrict__ bd2, const float* __restrict__ bg1,
    const float* __restrict__ bg2, const float* __restrict__ b2,
    float* __restrict__ out) {
  __shared__ _Float16 tA[4][16 * TSTRIDE];   // matmul staging tile (ch x neigh)
  __shared__ _Float16 tVP[4][16 * TSTRIDE];  // v + pos_enc tile
  __shared__ float sS[4][64 * 17];           // attention logits (padded)
  __shared__ float sQ[4][64];                // q vector of this point
  __shared__ float sR[4][64];                // attention result vector

  const int lane = threadIdx.x & 31, w = threadIdx.x >> 5;
  const int nl = lane & 15, h = lane >> 4;
  const int b = blockIdx.z;
  const int n = blockIdx.x * 4 + w;

  const int gi = knn[((size_t)(b * NN + n)) * KK + nl];
  const float rx = pos[(b * 3 + 0) * NN + n] - pos[(b * 3 + 0) * NN + gi];
  const float ry = pos[(b * 3 + 1) * NN + n] - pos[(b * 3 + 1) * NN + gi];
  const float rz = pos[(b * 3 + 2) * NN + n] - pos[(b * 3 + 2) * NN + gi];

  sQ[w][lane]      = qg[(b * DD + lane) * NN + n];
  sQ[w][lane + 32] = qg[(b * DD + lane + 32) * NN + n];

  // pe1 = lrelu(bn(Wd1 * rel)) — tiny 64x3 matvec done on VALU.
  // lane handles neighbor column nl, channels [32h, 32h+32).
#pragma unroll
  for (int c = 0; c < 32; c++) {
    int ch = 32 * h + c;
    float v = fWd1s[ch * 3 + 0] * rx + fWd1s[ch * 3 + 1] * ry +
              fWd1s[ch * 3 + 2] * rz + bd1[ch];
    tA[w][nl * TSTRIDE + ch] = (_Float16)lrelu(v);
  }
  __syncthreads();

  // pos_enc = lrelu(bn(Wd2 * pe1)); build a-tile and (v+pos_enc)-tile.
  {
    v16h bfr0 = ldB_lds(tA[w], 0, nl, h);
    v16h bfr1 = ldB_lds(tA[w], 1, nl, h);
    const size_t kvoff = ((size_t)(b * NN) + gi) * DD;
#pragma unroll
    for (int rb = 0; rb < 4; rb++) {
      v8f acc = {};
      acc = wmma_f16(ldA(hWd2, rb, 0, nl, h), bfr0, acc);
      acc = wmma_f16(ldA(hWd2, rb, 1, nl, h), bfr1, acc);
      v8h kv = *(const v8h*)(kf + kvoff + 16 * rb + 8 * h);
      v8h vv = *(const v8h*)(vf + kvoff + 16 * rb + 8 * h);
#pragma unroll
      for (int r = 0; r < 8; r++) {
        int m = 16 * rb + r + 8 * h;
        float pe = lrelu(acc[r] + bd2[m]);
        float av = sQ[w][m] - (float)kv[r] + pe;   // a = q - k + pos_enc
        float vp = (float)vv[r] + pe;              // v + pos_enc
        tA[w][nl * TSTRIDE + m]  = (_Float16)av;   // DS in-order: safe overwrite
        tVP[w][nl * TSTRIDE + m] = (_Float16)vp;
      }
    }
  }
  __syncthreads();

  // a = lrelu(bn(Wg1 * a))
  {
    v16h bfr0 = ldB_lds(tA[w], 0, nl, h);
    v16h bfr1 = ldB_lds(tA[w], 1, nl, h);
#pragma unroll
    for (int rb = 0; rb < 4; rb++) {
      v8f acc = {};
      acc = wmma_f16(ldA(hWg1, rb, 0, nl, h), bfr0, acc);
      acc = wmma_f16(ldA(hWg1, rb, 1, nl, h), bfr1, acc);
#pragma unroll
      for (int r = 0; r < 8; r++) {
        int m = 16 * rb + r + 8 * h;
        tA[w][nl * TSTRIDE + m] = (_Float16)lrelu(acc[r] + bg1[m]);
      }
    }
  }
  __syncthreads();

  // a = lrelu(bn(Wg2 * a)) / sqrt(N) -> logits in LDS (f32)
  {
    v16h bfr0 = ldB_lds(tA[w], 0, nl, h);
    v16h bfr1 = ldB_lds(tA[w], 1, nl, h);
    const float scale = 0.011048543456039805f;  // 1/sqrt(8192)
#pragma unroll
    for (int rb = 0; rb < 4; rb++) {
      v8f acc = {};
      acc = wmma_f16(ldA(hWg2, rb, 0, nl, h), bfr0, acc);
      acc = wmma_f16(ldA(hWg2, rb, 1, nl, h), bfr1, acc);
#pragma unroll
      for (int r = 0; r < 8; r++) {
        int m = 16 * rb + r + 8 * h;
        sS[w][m * 17 + nl] = lrelu(acc[r] + bg2[m]) * scale;
      }
    }
  }
  __syncthreads();

  // softmax over the 16 neighbors per channel; weighted sum of (v+pos_enc).
#pragma unroll
  for (int cc = 0; cc < 2; cc++) {
    int m = lane + 32 * cc;
    float mx = -3.0e38f;
#pragma unroll
    for (int j = 0; j < KK; j++) mx = fmaxf(mx, sS[w][m * 17 + j]);
    float sum = 0.f, res = 0.f;
#pragma unroll
    for (int j = 0; j < KK; j++) {
      float e = __expf(sS[w][m * 17 + j] - mx);
      sum += e;
      res += e * (float)tVP[w][j * TSTRIDE + m];
    }
    sR[w][m] = res / sum;
  }
  __syncthreads();

  // out = lrelu(bn(W2 * res)) + feats
#pragma unroll
  for (int cc = 0; cc < 2; cc++) {
    int mo = lane + 32 * cc;
    float acc = b2[mo];
    for (int mi = 0; mi < 64; mi++) acc += fW2s[mo * 64 + mi] * sR[w][mi];
    out[(b * DD + mo) * NN + n] = lrelu(acc) + feats[(b * DD + mo) * NN + n];
  }
}

// ---------------------------------------------------------------------------
extern "C" void kernel_launch(void* const* d_in, const int* in_sizes, int n_in,
                              void* d_out, int out_size, void* d_ws,
                              size_t ws_size, hipStream_t stream) {
  const float* feats = (const float*)d_in[0];
  const float* pos   = (const float*)d_in[1];
  // d_in[2] = k (always 16 here)
  const float* W1  = (const float*)d_in[3];
  const float* g1  = (const float*)d_in[4];
  const float* b1  = (const float*)d_in[5];
  const float* Wq  = (const float*)d_in[6];
  const float* Wk  = (const float*)d_in[7];
  const float* Wv  = (const float*)d_in[8];
  const float* Wd1 = (const float*)d_in[9];
  const float* gd1 = (const float*)d_in[10];
  const float* bd1 = (const float*)d_in[11];
  const float* Wd2 = (const float*)d_in[12];
  const float* gd2 = (const float*)d_in[13];
  const float* bd2 = (const float*)d_in[14];
  const float* Wg1 = (const float*)d_in[15];
  const float* gg1 = (const float*)d_in[16];
  const float* bg1 = (const float*)d_in[17];
  const float* Wg2 = (const float*)d_in[18];
  const float* gg2 = (const float*)d_in[19];
  const float* bg2 = (const float*)d_in[20];
  const float* W2  = (const float*)d_in[21];
  const float* g2  = (const float*)d_in[22];
  const float* b2  = (const float*)d_in[23];
  float* out = (float*)d_out;

  // Workspace layout (bytes), all 16B-aligned. Total ~9.1 MB.
  char* ws = (char*)d_ws;
  _Float16* hW1  = (_Float16*)(ws + 0);
  _Float16* hWq  = (_Float16*)(ws + 8192);
  _Float16* hWk  = (_Float16*)(ws + 16384);
  _Float16* hWv  = (_Float16*)(ws + 24576);
  _Float16* hWd2 = (_Float16*)(ws + 32768);
  _Float16* hWg1 = (_Float16*)(ws + 40960);
  _Float16* hWg2 = (_Float16*)(ws + 49152);
  float*    fWd1s = (float*)(ws + 57344);                 // 192 f32
  float*    fW2s  = (float*)(ws + 58368);                 // 4096 f32
  float*    qbuf  = (float*)(ws + 74752);                 // B*64*N f32 = 4 MB
  _Float16* kbuf  = (_Float16*)(ws + 74752 + 4194304);    // B*N*64 f16 = 2 MB
  _Float16* vbuf  = (_Float16*)(ws + 74752 + 6291456);    // B*N*64 f16 = 2 MB
  int*      ibuf  = (int*)(ws + 74752 + 8388608);         // B*N*16 i32 = 1 MB

  prep_kernel<<<dim3(16), 256, 0, stream>>>(
      W1, g1, Wq, Wk, Wv, Wd1, gd1, Wd2, gd2, Wg1, gg1, Wg2, gg2, W2, g2,
      hW1, hWq, hWk, hWv, hWd2, hWg1, hWg2, fWd1s, fW2s);

  qkv_kernel<<<dim3(NN / 64, 1, BB), 128, 0, stream>>>(
      feats, b1, hW1, hWq, hWk, hWv, qbuf, kbuf, vbuf);

  knn_kernel<<<dim3(NN / 256, 1, BB), 256, 0, stream>>>(pos, ibuf);

  attn_kernel<<<dim3(NN / 4, 1, BB), 128, 0, stream>>>(
      feats, pos, qbuf, kbuf, vbuf, ibuf, hWd2, hWg1, hWg2, fWd1s, fW2s,
      bd1, bd2, bg1, bg2, b2, out);
}
